// EmbedMatcher_1142461301175
// MI455X (gfx1250) — matrix operand order repaired and checked by reading, not compiled, not run on hardware
//
#include <hip/hip_runtime.h>
#include <hip/hip_bf16.h>

// ---------------------------------------------------------------------------
// EmbedMatcher (GMatching) forward, factorized for MI455X (gfx1250).
//   D=256, D2=512, BATCH=1024, FEW=5, K=128 neighbors.
// Attention scores are rank-1 in the projections and the weighted aggregation
// commutes with the linear maps, so the neighbor encoder becomes: scalar
// gathers + softmax + weighted embedding sums + one bf16 WMMA GEMM.
// GEMM waves compute 16x64 tiles (A-fragment reused across 4 WMMAs), with a
// sched_barrier pinning fragment loads ahead of the WMMA group so the waits
// stay partial (load/WMMA overlap within the wave).
// ---------------------------------------------------------------------------

#define DD    256
#define DD2   512
#define KNEI  128
#define NBATCH 1024
#define NFEW  5
#define NHEADS (2*NBATCH + 2*NFEW)   // 2058
#define MP_HEADS 2064                // padded to /16
#define ROWS_SE (NBATCH + NFEW)      // 1029
#define MP_SE 1040                   // padded to /16
#define NENT 100001
#define NREL 501

typedef __attribute__((ext_vector_type(16))) __bf16 v16bf;
typedef __attribute__((ext_vector_type(8)))  float  v8f;

__device__ __forceinline__ float lrelu(float x) { return x >= 0.f ? x : 0.2f * x; }

// --------------------------- WMMA fragment loads ---------------------------
// A: 16x32 bf16 tile from row-major (ld = K). Layout used here (contiguous k): lane L holds row
// m=L&15; element e -> k = (L>>4)*16 + e.
__device__ __forceinline__ v16bf load_a_frag(const __bf16* A, int ld, int row0, int k0, int lane) {
  int m = row0 + (lane & 15);
  int half = (lane >> 4) & 1;
  const __bf16* p = A + (size_t)m * ld + k0 + half * 16;
  v16bf f;
#pragma unroll
  for (int e = 0; e < 16; ++e) f[e] = p[e];
  return f;
}
// B operand (KxN) taken from N x K row-major weight W: lane L holds col
// n=L&15; lanes 0-15 carry K=k0..k0+15, lanes 16-31 carry K=k0+16..k0+31.
__device__ __forceinline__ v16bf load_b_frag(const __bf16* B, int ld, int col0, int k0, int lane) {
  int n = col0 + (lane & 15);
  int half = (lane >> 4) & 1;
  const __bf16* p = B + (size_t)n * ld + k0 + half * 16;
  v16bf f;
#pragma unroll
  for (int e = 0; e < 16; ++e) f[e] = p[e];
  return f;
}

// ------------------------ generic NT bf16 WMMA GEMM ------------------------
// C(MpxN,f32) = A(MpxK) @ B(NxK)^T  [+bias][relu][+resid]; optional bf16 copy.
// Each wave owns a 16x64 tile: 1 A-frag feeds 4 WMMAs per K-step.
__global__ void wmma_gemm_nt(const __bf16* __restrict__ A, const __bf16* __restrict__ B,
                             float* __restrict__ C, __bf16* __restrict__ Cb,
                             const float* __restrict__ bias,
                             const float* __restrict__ resid, int resid_ld,
                             int Mp, int N, int Kd, int relu) {
  int lane = threadIdx.x & 31;
  int wave = threadIdx.x >> 5;
  int ntn = N >> 6;                              // 64-wide N tiles
  int total = (Mp >> 4) * ntn;
  int t = blockIdx.x * (blockDim.x >> 5) + wave; // wave-uniform: EXEC stays full
  if (t >= total) return;
  int mt = t / ntn, nt = t - mt * ntn;
  int m0 = mt << 4, n0 = nt << 6;
  v8f acc[4] = {};
  for (int k0 = 0; k0 < Kd; k0 += 32) {
    v16bf a  = load_a_frag(A, Kd, m0, k0, lane);
    v16bf b0 = load_b_frag(B, Kd, n0,      k0, lane);
    v16bf b1 = load_b_frag(B, Kd, n0 + 16, k0, lane);
    v16bf b2 = load_b_frag(B, Kd, n0 + 32, k0, lane);
    v16bf b3 = load_b_frag(B, Kd, n0 + 48, k0, lane);
    // Keep all fragment loads issued before the WMMA group so the per-WMMA
    // waits stay partial (loadcnt 3/2/1/0) instead of full serialization.
    __builtin_amdgcn_sched_barrier(0);
    acc[0] = __builtin_amdgcn_wmma_f32_16x16x32_bf16(false, a, false, b0, (short)0, acc[0], false, false);
    acc[1] = __builtin_amdgcn_wmma_f32_16x16x32_bf16(false, a, false, b1, (short)0, acc[1], false, false);
    acc[2] = __builtin_amdgcn_wmma_f32_16x16x32_bf16(false, a, false, b2, (short)0, acc[2], false, false);
    acc[3] = __builtin_amdgcn_wmma_f32_16x16x32_bf16(false, a, false, b3, (short)0, acc[3], false, false);
  }
  int mbase = m0 + ((lane >> 4) << 3);
#pragma unroll
  for (int s = 0; s < 4; ++s) {
    int n = n0 + s * 16 + (lane & 15);
    float bv = bias ? bias[n] : 0.f;
#pragma unroll
    for (int v = 0; v < 8; ++v) {
      int m = mbase + v;
      float val = acc[s][v] + bv;
      if (relu) val = fmaxf(val, 0.f);
      if (resid) val += resid[(size_t)m * resid_ld + n];
      if (C)  C[(size_t)m * N + n] = val;
      if (Cb) Cb[(size_t)m * N + n] = (__bf16)val;
    }
  }
}

// ------------------- tiny rank-1 / constant precompute ---------------------
__global__ void precompute_small(const float* __restrict__ Ww, const float* __restrict__ bw,
                                 const float* __restrict__ Wp, const float* __restrict__ bp,
                                 const float* __restrict__ Ww2, const float* __restrict__ bw2,
                                 const float* __restrict__ Wq, const float* __restrict__ bq,
                                 const float* __restrict__ rel_emb,
                                 float* vp_h, float* vp_r, float* vq_h, float* vq_r, float* vq_e,
                                 float* cb, float* consts, float* sr, float* tr) {
  __shared__ float u[DD], cbs[DD], vprs[DD], vqrs[DD];
  int t = threadIdx.x;  // 256
  float acc = 0.f;
  for (int m = 0; m < DD; ++m) acc += Ww2[t * DD2 + m] * bw[m];
  cbs[t] = acc + bw2[t];
  cb[t] = cbs[t];
  acc = 0.f;                                   // u = Ww2a^T @ Wq
  for (int d = 0; d < DD; ++d) acc += Ww2[d * DD2 + t] * Wq[d];
  u[t] = acc;
  __syncthreads();
  float a1 = 0, a2 = 0, a3 = 0, a4 = 0, a5 = 0;
  for (int d = 0; d < DD; ++d) {
    float wd = Wp[d];
    a1 += Ww[d * DD2 + t] * wd;                // vp_h = Wwa^T@Wp
    a2 += Ww[d * DD2 + DD + t] * wd;           // vp_r = Wwb^T@Wp
    a3 += Ww2[d * DD2 + DD + t] * Wq[d];       // vq_e = Ww2b^T@Wq
  }
  for (int m = 0; m < DD; ++m) {
    a4 += Ww[m * DD2 + t] * u[m];              // vq_h = Wwa^T@u
    a5 += Ww[m * DD2 + DD + t] * u[m];         // vq_r = Wwb^T@u
  }
  vp_h[t] = a1; vp_r[t] = a2; vq_e[t] = a3; vq_h[t] = a4; vq_r[t] = a5;
  vprs[t] = a2; vqrs[t] = a5;
  __syncthreads();
  if (t == 0) {
    float c1 = bp[0]; for (int d = 0; d < DD; ++d) c1 += Wp[d] * bw[d];
    float c2 = bq[0]; for (int d = 0; d < DD; ++d) c2 += Wq[d] * cbs[d];
    consts[0] = c1; consts[1] = c2;
  }
  for (int r = t; r < NREL; r += 256) {        // per-relation score scalars
    float s = 0.f, q = 0.f;
    for (int d = 0; d < DD; ++d) { float re = rel_emb[(size_t)r * DD + d]; s += re * vprs[d]; q += re * vqrs[d]; }
    sr[r] = s; tr[r] = q;
  }
}

// M = [Ww2a@Wwa | Ww2a@Wwb | Ww2b] as bf16 (256 x 768)
__global__ void build_M(const float* __restrict__ Ww, const float* __restrict__ Ww2,
                        __bf16* __restrict__ Mb) {
  int idx = blockIdx.x * 256 + threadIdx.x;    // < 256*768
  int d = idx / 768, c = idx % 768;
  float val;
  if (c < 512) {
    float s = 0.f;
    for (int m = 0; m < DD; ++m) s += Ww2[d * DD2 + m] * Ww[m * DD2 + c];
    val = s;
  } else {
    val = Ww2[d * DD2 + DD + (c - 512)];
  }
  Mb[idx] = (__bf16)val;
}

// out[row] = dot(emb[row], vec)  (wave per row, coalesced 32B/lane)
__global__ void dot_table(const float* __restrict__ emb, const float* __restrict__ vec,
                          float* __restrict__ out, int nrows) {
  int wave = threadIdx.x >> 5, lane = threadIdx.x & 31;
  int row = blockIdx.x * 8 + wave;
  if (row >= nrows) return;
  const float* p = emb + (size_t)row * DD + lane * 8;
  float s = 0.f;
#pragma unroll
  for (int j = 0; j < 8; ++j) s += p[j] * vec[lane * 8 + j];
  for (int off = 16; off; off >>= 1) s += __shfl_down(s, off);
  if (lane == 0) out[row] = s;
}

__global__ void f2bf(const float* __restrict__ in, __bf16* __restrict__ out, int n) {
  int i = blockIdx.x * 256 + threadIdx.x;
  if (i < n) out[i] = (__bf16)in[i];
}

// ------------------- neighbor encoder: scores + gathers --------------------
__global__ void neighbor_heads(const int* __restrict__ ids, int stride, int off,
                               const int* __restrict__ conns,
                               const float* __restrict__ ent_emb, const float* __restrict__ rel_emb,
                               const float* __restrict__ vp_h, const float* __restrict__ vq_h,
                               const float* __restrict__ sr, const float* __restrict__ tr,
                               const float* __restrict__ te, const float* __restrict__ consts,
                               __bf16* __restrict__ Xb, float* __restrict__ W0arr,
                               float* __restrict__ Hrow, int row_base) {
  __shared__ float hrow[DD], red[DD];
  __shared__ float sc1[KNEI], sc2[KNEI], wv[KNEI];
  __shared__ int rl[KNEI], el[KNEI];
  __shared__ float scal[8];
  int n = blockIdx.x, t = threadIdx.x;
  int row = row_base + n;
  int hid = ids[n * stride + off];
  hrow[t] = ent_emb[(size_t)hid * DD + t];
  __syncthreads();
  red[t] = hrow[t] * vp_h[t];
  __syncthreads();
  for (int s = 128; s; s >>= 1) { if (t < s) red[t] += red[t + s]; __syncthreads(); }
  if (t == 0) scal[0] = red[0];
  __syncthreads();
  red[t] = hrow[t] * vq_h[t];
  __syncthreads();
  for (int s = 128; s; s >>= 1) { if (t < s) red[t] += red[t + s]; __syncthreads(); }
  if (t == 0) scal[1] = red[0];
  __syncthreads();
  float s_h = scal[0], t_h = scal[1], c1 = consts[0], c2 = consts[1];
  if (t < KNEI) {
    int r = conns[((size_t)n * KNEI + t) * 2];
    int e = conns[((size_t)n * KNEI + t) * 2 + 1];
    rl[t] = r; el[t] = e;
    sc1[t] = lrelu(s_h + sr[r] + c1);
    sc2[t] = lrelu(t_h + tr[r] + te[e] + c2);
  }
  __syncthreads();
  if (t < 64) red[t] = fmaxf(sc1[t], sc1[t + 64]);
  __syncthreads();
  for (int s = 32; s; s >>= 1) { if (t < s) red[t] = fmaxf(red[t], red[t + s]); __syncthreads(); }
  if (t == 0) scal[2] = red[0];
  __syncthreads();
  if (t < 64) red[t] = fmaxf(sc2[t], sc2[t + 64]);
  __syncthreads();
  for (int s = 32; s; s >>= 1) { if (t < s) red[t] = fmaxf(red[t], red[t + s]); __syncthreads(); }
  if (t == 0) scal[3] = red[0];
  __syncthreads();
  if (t < KNEI) { sc1[t] = __expf(sc1[t] - scal[2]); sc2[t] = __expf(sc2[t] - scal[3]); }
  __syncthreads();
  if (t < 64) red[t] = sc1[t] + sc1[t + 64];
  __syncthreads();
  for (int s = 32; s; s >>= 1) { if (t < s) red[t] += red[t + s]; __syncthreads(); }
  if (t == 0) scal[4] = red[0];
  __syncthreads();
  if (t < 64) red[t] = sc2[t] + sc2[t + 64];
  __syncthreads();
  for (int s = 32; s; s >>= 1) { if (t < s) red[t] += red[t + s]; __syncthreads(); }
  if (t == 0) scal[5] = red[0];
  __syncthreads();
  float inv = 1.f / (scal[4] * scal[5]);
  if (t < KNEI) wv[t] = sc1[t] * sc2[t] * inv;   // w_hr * w_hrt
  __syncthreads();
  if (t < 64) red[t] = wv[t] + wv[t + 64];
  __syncthreads();
  for (int s = 32; s; s >>= 1) { if (t < s) red[t] += red[t + s]; __syncthreads(); }
  if (t == 0) scal[6] = red[0];
  __syncthreads();
  float W0 = scal[6];
  float ur = 0.f, ue = 0.f;                      // weighted embedding sums (L2 gathers)
  for (int k = 0; k < KNEI; ++k) {
    float w = wv[k];
    ur += w * rel_emb[(size_t)rl[k] * DD + t];
    ue += w * ent_emb[(size_t)el[k] * DD + t];
  }
  __bf16* xr = Xb + (size_t)row * 768;
  xr[t]          = (__bf16)(W0 * hrow[t]);
  xr[DD + t]     = (__bf16)ur;
  xr[2 * DD + t] = (__bf16)ue;
  Hrow[(size_t)row * DD + t] = hrow[t];
  if (t == 0) W0arr[row] = W0;
}

// agg -> neighbor-encoder output -> scatter into support-encoder input
__global__ void ne_epilogue(const float* __restrict__ C1, const float* __restrict__ W0arr,
                            const float* __restrict__ Hrow, const float* __restrict__ cb,
                            float* __restrict__ xg, __bf16* __restrict__ xgb) {
  int i = blockIdx.x, d = threadIdx.x;
  float agg = C1[(size_t)i * DD + d] + W0arr[i] * cb[d];
  float h = Hrow[(size_t)i * DD + d];
  float o = 0.5f * (lrelu(agg + h) + lrelu(h * agg));
  int row, col;
  if (i < NBATCH)                { row = i;                         col = d; }
  else if (i < 2 * NBATCH)       { row = i - NBATCH;                col = DD + d; }
  else if (i < 2 * NBATCH + NFEW){ row = NBATCH + (i - 2 * NBATCH); col = d; }
  else                           { row = NBATCH + (i - 2 * NBATCH - NFEW); col = DD + d; }
  xg[(size_t)row * DD2 + col] = o;
  xgb[(size_t)row * DD2 + col] = (__bf16)o;
}

// GMatching layernorm: std with ddof=1, eps added to std
__global__ void layernorm_rows(const float* __restrict__ h, const float* __restrict__ g,
                               const float* __restrict__ b, float* __restrict__ out) {
  __shared__ float red[256], red2[256];
  int r = blockIdx.x, t = threadIdx.x;
  float x0 = h[(size_t)r * DD2 + t], x1 = h[(size_t)r * DD2 + 256 + t];
  red[t] = x0 + x1; red2[t] = x0 * x0 + x1 * x1;
  __syncthreads();
  for (int s = 128; s; s >>= 1) { if (t < s) { red[t] += red[t + s]; red2[t] += red2[t + s]; } __syncthreads(); }
  float mu = red[0] / 512.f;
  float var = (red2[0] - 512.f * mu * mu) / 511.f;
  float inv = 1.f / (sqrtf(fmaxf(var, 0.f)) + 1e-3f);
  out[(size_t)r * DD2 + t]       = g[t] * (x0 - mu) * inv + b[t];
  out[(size_t)r * DD2 + 256 + t] = g[256 + t] * (x1 - mu) * inv + b[256 + t];
}

// one LSTM step, batch=1, hidden=512 (gate order i,f,g,o); also emits S[t]
__global__ void lstm_step(const float* __restrict__ G, int trow,
                          const float* __restrict__ Wih, const float* __restrict__ Whh,
                          const float* __restrict__ bih, const float* __restrict__ bhh,
                          float* __restrict__ hbuf, float* __restrict__ cbuf,
                          float* __restrict__ S, int t_step) {
  __shared__ float xs[DD2], hs[DD2];
  int j = threadIdx.x;  // 512
  float xv = G[(size_t)trow * DD2 + j];
  xs[j] = xv;
  hs[j] = (t_step == 0) ? 0.f : hbuf[j];
  float cprev = (t_step == 0) ? 0.f : cbuf[j];
  __syncthreads();
  float gate[4];
#pragma unroll
  for (int gi = 0; gi < 4; ++gi) {
    int rowi = gi * DD2 + j;
    float a = bih[rowi] + bhh[rowi];
    const float* wi = Wih + (size_t)rowi * DD2;
    const float* wh = Whh + (size_t)rowi * DD2;
    for (int c = 0; c < DD2; ++c) a += xs[c] * wi[c] + hs[c] * wh[c];
    gate[gi] = a;
  }
  float ig = 1.f / (1.f + __expf(-gate[0]));
  float fg = 1.f / (1.f + __expf(-gate[1]));
  float gg = tanhf(gate[2]);
  float og = 1.f / (1.f + __expf(-gate[3]));
  float c = fg * cprev + ig * gg;
  float hn = og * tanhf(c);
  hbuf[j] = hn; cbuf[j] = c;
  S[t_step * DD2 + j] = 0.5f * (xv + hn);       // (support_g + enc)/2
}

// out[n] = scale * sum_f softmax(q.S^T)_f * (q.S_f)   (since q.agg = sum wei*logit)
__global__ void final_score(const float* __restrict__ G, const float* __restrict__ S,
                            float* __restrict__ out) {
  __shared__ float red[256];
  __shared__ float lf[NFEW];
  int n = blockIdx.x, t = threadIdx.x;
  const float* q = G + (size_t)n * DD2;
  float q0 = q[t], q1 = q[256 + t];
  for (int f = 0; f < NFEW; ++f) {
    red[t] = q0 * S[f * DD2 + t] + q1 * S[f * DD2 + 256 + t];
    __syncthreads();
    for (int s = 128; s; s >>= 1) { if (t < s) red[t] += red[t + s]; __syncthreads(); }
    if (t == 0) lf[f] = red[0];
    __syncthreads();
  }
  if (t == 0) {
    float m = lf[0];
    for (int f = 1; f < NFEW; ++f) m = fmaxf(m, lf[f]);
    float ss = 0.f, acc = 0.f;
    for (int f = 0; f < NFEW; ++f) { float e = __expf(lf[f] - m); ss += e; acc += e * lf[f]; }
    out[n] = (acc / ss) * 0.044194173824159216f;  // 512^-0.5
  }
}

// ---------------------------------------------------------------------------
extern "C" void kernel_launch(void* const* d_in, const int* in_sizes, int n_in,
                              void* d_out, int out_size, void* d_ws, size_t ws_size,
                              hipStream_t stream) {
  (void)in_sizes; (void)n_in; (void)out_size; (void)ws_size;
  const int*   query   = (const int*)d_in[1];
  const int*   support = (const int*)d_in[2];
  const int*   qlc     = (const int*)d_in[3];
  const int*   qrc     = (const int*)d_in[4];
  const int*   slc     = (const int*)d_in[5];
  const int*   srcn    = (const int*)d_in[6];
  const float* ent_emb = (const float*)d_in[7];
  const float* rel_emb = (const float*)d_in[8];
  const float* Ww  = (const float*)d_in[9];
  const float* bw  = (const float*)d_in[10];
  const float* Wp  = (const float*)d_in[11];
  const float* bp  = (const float*)d_in[12];
  const float* Ww2 = (const float*)d_in[13];
  const float* bw2 = (const float*)d_in[14];
  const float* Wq  = (const float*)d_in[15];
  const float* bq  = (const float*)d_in[16];
  const float* p1W = (const float*)d_in[17];
  const float* p1b = (const float*)d_in[18];
  const float* p2W = (const float*)d_in[19];
  const float* p2b = (const float*)d_in[20];
  const float* lng = (const float*)d_in[21];
  const float* lnb = (const float*)d_in[22];
  const float* Wih = (const float*)d_in[23];
  const float* Whh = (const float*)d_in[24];
  const float* bih = (const float*)d_in[25];
  const float* bhh = (const float*)d_in[26];

  char* base = (char*)d_ws; size_t off = 0;
  auto alloc = [&](size_t bytes) -> void* {
    void* p = base + off; off = (off + bytes + 255) & ~(size_t)255; return p;
  };
  float* vp_h   = (float*)alloc(DD * 4);
  float* vp_r   = (float*)alloc(DD * 4);
  float* vq_h   = (float*)alloc(DD * 4);
  float* vq_r   = (float*)alloc(DD * 4);
  float* vq_e   = (float*)alloc(DD * 4);
  float* cb     = (float*)alloc(DD * 4);
  float* consts = (float*)alloc(8 * 4);
  float* sr     = (float*)alloc(512 * 4);
  float* tr     = (float*)alloc(512 * 4);
  float* te     = (float*)alloc((size_t)100352 * 4);
  float* W0arr  = (float*)alloc(MP_HEADS * 4);
  float* Hrow   = (float*)alloc((size_t)MP_HEADS * DD * 4);
  float* C1     = (float*)alloc((size_t)MP_HEADS * DD * 4);
  float* xg     = (float*)alloc((size_t)MP_SE * DD2 * 4);
  float* hmat   = (float*)alloc((size_t)MP_SE * DD2 * 4);
  float* Gm     = (float*)alloc((size_t)MP_SE * DD2 * 4);
  float* Sm     = (float*)alloc(NFEW * DD2 * 4);
  float* hbuf   = (float*)alloc(DD2 * 4);
  float* cbuf   = (float*)alloc(DD2 * 4);
  __bf16* Mb   = (__bf16*)alloc((size_t)DD * 768 * 2);
  __bf16* Xb   = (__bf16*)alloc((size_t)MP_HEADS * 768 * 2);
  __bf16* xgb  = (__bf16*)alloc((size_t)MP_SE * DD2 * 2);
  __bf16* p1Wb = (__bf16*)alloc((size_t)1024 * 512 * 2);
  __bf16* p2Wb = (__bf16*)alloc((size_t)512 * 1024 * 2);
  __bf16* h1b  = (__bf16*)alloc((size_t)MP_SE * 1024 * 2);

  // Stage 1: rank-1 vectors, constants, per-relation scalars, fused M matrix
  precompute_small<<<1, 256, 0, stream>>>(Ww, bw, Wp, bp, Ww2, bw2, Wq, bq, rel_emb,
                                          vp_h, vp_r, vq_h, vq_r, vq_e, cb, consts, sr, tr);
  build_M<<<768, 256, 0, stream>>>(Ww, Ww2, Mb);
  dot_table<<<(NENT + 7) / 8, 256, 0, stream>>>(ent_emb, vq_e, te, NENT);
  f2bf<<<(1024 * 512 + 255) / 256, 256, 0, stream>>>(p1W, p1Wb, 1024 * 512);
  f2bf<<<(512 * 1024 + 255) / 256, 256, 0, stream>>>(p2W, p2Wb, 512 * 1024);

  // Stage 2: neighbor attention (softmax + weighted gathers), 4 head sets
  neighbor_heads<<<NBATCH, 256, 0, stream>>>(query, 2, 0, qlc, ent_emb, rel_emb, vp_h, vq_h,
                                             sr, tr, te, consts, Xb, W0arr, Hrow, 0);
  neighbor_heads<<<NBATCH, 256, 0, stream>>>(query, 2, 1, qrc, ent_emb, rel_emb, vp_h, vq_h,
                                             sr, tr, te, consts, Xb, W0arr, Hrow, NBATCH);
  neighbor_heads<<<NFEW, 256, 0, stream>>>(support, 2, 0, slc, ent_emb, rel_emb, vp_h, vq_h,
                                           sr, tr, te, consts, Xb, W0arr, Hrow, 2 * NBATCH);
  neighbor_heads<<<NFEW, 256, 0, stream>>>(support, 2, 1, srcn, ent_emb, rel_emb, vp_h, vq_h,
                                           sr, tr, te, consts, Xb, W0arr, Hrow, 2 * NBATCH + NFEW);

  // Stage 3: agg = X @ M^T  (WMMA bf16), then neighbor-encoder epilogue
  {
    int tiles = (MP_HEADS / 16) * (DD / 64);
    wmma_gemm_nt<<<(tiles + 7) / 8, 256, 0, stream>>>(Xb, Mb, C1, nullptr, nullptr,
                                                      nullptr, 0, MP_HEADS, DD, 768, 0);
  }
  ne_epilogue<<<NHEADS, 256, 0, stream>>>(C1, W0arr, Hrow, cb, xg, xgb);

  // Stage 4: support encoder: relu(x@W1^T+b1)@W2^T + b2 + x, then layernorm
  {
    int tiles = (MP_SE / 16) * (1024 / 64);
    wmma_gemm_nt<<<(tiles + 7) / 8, 256, 0, stream>>>(xgb, p1Wb, nullptr, h1b, p1b,
                                                      nullptr, 0, MP_SE, 1024, 512, 1);
  }
  {
    int tiles = (MP_SE / 16) * (DD2 / 64);
    wmma_gemm_nt<<<(tiles + 7) / 8, 256, 0, stream>>>(h1b, p2Wb, hmat, nullptr, p2b,
                                                      xg, DD2, MP_SE, DD2, 1024, 0);
  }
  layernorm_rows<<<ROWS_SE, 256, 0, stream>>>(hmat, lng, lnb, Gm);

  // Stage 5: 5-step batch-1 LSTM over support rows, emits S = (sg + enc)/2
  for (int t = 0; t < NFEW; ++t)
    lstm_step<<<1, 512, 0, stream>>>(Gm, NBATCH + t, Wih, Whh, bih, bhh, hbuf, cbuf, Sm, t);

  // Stage 6: matching score (softmax-weighted self-dot of logits)
  final_score<<<NBATCH, 256, 0, stream>>>(Gm, Sm, (float*)d_out);
}
